// HardNegTripletMarginLoss_57509612094151
// MI455X (gfx1250) — matrix operand order repaired and checked by reading, not compiled, not run on hardware
//
#include <hip/hip_runtime.h>

typedef __attribute__((ext_vector_type(16))) _Float16 v16h;
typedef __attribute__((ext_vector_type(8)))  _Float16 v8h;
typedef __attribute__((ext_vector_type(4)))  _Float16 v4h;
typedef __attribute__((ext_vector_type(8)))  float    v8f;

#define N_SAMP 8192
#define DIMF   128
#define MARGIN_V 0.05f
#define FMAXV  3.402823466e+38f
// bit patterns of the reduction sentinels (valid signed-int ordering for d2 >= 0)
#define PMAX_SENT 0xBF800000   /* asint(-1.0f): loses every signed-int max vs d2>=0 */
#define NMIN_SENT 0x7F7FFFFF   /* asint(FLT_MAX): loses every signed-int min vs d2  */

// ---------------------------------------------------------------------------
// Kernel 1: row-wise L2 normalize, emit f16 copy of x and sq[i] = sum(x*x).
// One wave (32 lanes) per row, 4 floats per lane. Also zero the accumulators.
// ---------------------------------------------------------------------------
__global__ void __launch_bounds__(256)
normalize_kernel(const float* __restrict__ emb, _Float16* __restrict__ xh,
                 float* __restrict__ sq, float* __restrict__ accum) {
  if (blockIdx.x == 0 && threadIdx.x == 0) { accum[0] = 0.0f; accum[1] = 0.0f; }
  const int row = blockIdx.x * 8 + (threadIdx.x >> 5);
  const int l   = threadIdx.x & 31;
  const float4 v = ((const float4*)(emb + (size_t)row * DIMF))[l];
  float ss = v.x*v.x + v.y*v.y + v.z*v.z + v.w*v.w;
  #pragma unroll
  for (int m = 16; m >= 1; m >>= 1) ss += __shfl_xor(ss, m, 32);
  const float inv = 1.0f / fmaxf(sqrtf(ss), 1e-12f);
  const float x0 = v.x*inv, x1 = v.y*inv, x2 = v.z*inv, x3 = v.w*inv;
  float s2 = x0*x0 + x1*x1 + x2*x2 + x3*x3;
  #pragma unroll
  for (int m = 16; m >= 1; m >>= 1) s2 += __shfl_xor(s2, m, 32);
  v4h h; h[0] = (_Float16)x0; h[1] = (_Float16)x1; h[2] = (_Float16)x2; h[3] = (_Float16)x3;
  *(v4h*)(xh + (size_t)row * DIMF + l * 4) = h;
  if (l == 0) sq[row] = s2;
}

// ---------------------------------------------------------------------------
// Kernel 2: each block owns 16 anchor rows; 8 waves sweep the 512 column tiles.
// Double-buffered B fragments: tile t+1's 8x b128 loads issue before tile t's
// 4x v_wmma_f32_16x16x32_f16, hiding load latency. Hardest-pos / hardest-neg
// running reductions carried as int bit patterns (v_max_i32 / v_min_i32, no
// float canonicalization); sqrt deferred to the per-anchor finalize.
// ---------------------------------------------------------------------------
__global__ void __launch_bounds__(256)
triplet_tiles_kernel(const _Float16* __restrict__ xh,
                     const float*   __restrict__ sq,
                     const int*     __restrict__ labels,
                     float*         __restrict__ accum) {
  __shared__ int   s_pmax[16];
  __shared__ int   s_nmin[16];
  __shared__ float s_per[16];

  const int i0  = blockIdx.x << 4;     // anchor strip base
  const int wid = threadIdx.x >> 5;    // wave 0..7
  const int l   = threadIdx.x & 31;    // lane
  const int m   = l & 15;
  const int hi  = l >> 4;

  if (threadIdx.x < 16) {
    s_pmax[threadIdx.x] = (int)PMAX_SENT;
    s_nmin[threadIdx.x] = (int)NMIN_SENT;
  }
  __syncthreads();

  // Preload A fragments for K = 0..127 (contiguous K order, matching B):
  // lane holds row i0+m; halves 0..15 -> K = k0 + 16*hi + t
  v16h afrag[4];
  const _Float16* arow = xh + (size_t)(i0 + m) * DIMF;
  #pragma unroll
  for (int kk = 0; kk < 4; ++kk) {
    const int k0 = kk * 32;
    afrag[kk] = *(const v16h*)(arow + k0 + hi * 16);
  }

  // Row (anchor) metadata for this lane's 8 C-matrix elements (M = v + 8*hi)
  int rlab[8]; float rsq[8];
  #pragma unroll
  for (int v = 0; v < 8; ++v) {
    const int ir = i0 + v + 8 * hi;
    rlab[v] = labels[ir];
    rsq[v]  = sq[ir];
  }

  int ipmax[8], inmin[8];
  #pragma unroll
  for (int v = 0; v < 8; ++v) { ipmax[v] = (int)PMAX_SENT; inmin[v] = (int)NMIN_SENT; }

  const int kb = hi * 16;   // B fragment K-offset per lane half (B is 32x16, lane = column)

  // --- software-pipelined column-tile loop ---------------------------------
  v16h bcur[4]; int jlab; float jsq; int jcol;
  {
    jcol = (wid << 4) + m;
    const _Float16* brow = xh + (size_t)jcol * DIMF;
    #pragma unroll
    for (int kk = 0; kk < 4; ++kk) bcur[kk] = *(const v16h*)(brow + kk * 32 + kb);
    jlab = labels[jcol];
    jsq  = sq[jcol];
  }

  for (int t = 0; t < 64; ++t) {
    // issue next tile's loads first (branchless wrap; overlaps with WMMAs)
    const int tn = (t + 1) & 63;
    const int njcol = ((wid + tn * 8) << 4) + m;
    const _Float16* nrow = xh + (size_t)njcol * DIMF;
    v16h bnxt[4];
    #pragma unroll
    for (int kk = 0; kk < 4; ++kk) bnxt[kk] = *(const v16h*)(nrow + kk * 32 + kb);
    const int   nlab = labels[njcol];
    const float nsq  = sq[njcol];
    // prefetch two tiles ahead into near caches
    const int tp = (t + 2) & 63;
    __builtin_prefetch(xh + (size_t)(((wid + tp * 8) << 4) + m) * DIMF, 0, 3);

    v8f acc = {};
    #pragma unroll
    for (int kk = 0; kk < 4; ++kk) {
      acc = __builtin_amdgcn_wmma_f32_16x16x32_f16(
          /*neg_a=*/false, afrag[kk], /*neg_b=*/false, bcur[kk],
          /*c_mod=*/(short)0, acc, /*reuse_a=*/false, /*reuse_b=*/false);
    }

    #pragma unroll
    for (int v = 0; v < 8; ++v) {
      const float d2 = fmaxf(rsq[v] + jsq - 2.0f * acc[v], 0.0f);
      const int  id2  = __float_as_int(d2);           // >= 0: int order == float order
      const bool same = (rlab[v] == jlab);
      const bool diag = ((i0 + v + 8 * hi) == jcol);
      ipmax[v] = max(ipmax[v], (same && !diag) ? id2 : (int)PMAX_SENT);
      inmin[v] = min(inmin[v], same ? (int)NMIN_SENT : id2);
    }

    // rotate double buffer
    #pragma unroll
    for (int kk = 0; kk < 4; ++kk) bcur[kk] = bnxt[kk];
    jlab = nlab; jsq = nsq; jcol = njcol;
  }

  // Cross-lane reduce within each half (lanes 0-15 -> anchor v, 16-31 -> v+8),
  // then LDS int atomics combine the 8 waves.
  #pragma unroll
  for (int v = 0; v < 8; ++v) {
    int p = ipmax[v], q = inmin[v];
    #pragma unroll
    for (int mk = 1; mk < 16; mk <<= 1) {
      p = max(p, __shfl_xor(p, mk, 32));
      q = min(q, __shfl_xor(q, mk, 32));
    }
    if (m == 0) {
      atomicMax(&s_pmax[v + 8 * hi], p);
      atomicMin(&s_nmin[v + 8 * hi], q);
    }
  }
  __syncthreads();

  if (threadIdx.x < 16) {
    const float pm = __int_as_float(s_pmax[threadIdx.x]);
    const float nm = __int_as_float(s_nmin[threadIdx.x]);
    const bool valid = (pm >= 0.0f) && (nm < FMAXV);
    const float per = fmaxf(sqrtf(fmaxf(pm, 0.0f)) - sqrtf(fmaxf(nm, 0.0f)) + MARGIN_V, 0.0f);
    s_per[threadIdx.x] = valid ? per : 0.0f;
  }
  __syncthreads();
  if (threadIdx.x == 0) {
    float sum = 0.0f, cnt = 0.0f;
    #pragma unroll
    for (int t = 0; t < 16; ++t) { sum += s_per[t]; cnt += (s_per[t] > 0.0f) ? 1.0f : 0.0f; }
    atomicAdd(&accum[0], sum);
    atomicAdd(&accum[1], cnt);
  }
}

// ---------------------------------------------------------------------------
// Kernel 3: AvgNonZeroReducer finalize.
// ---------------------------------------------------------------------------
__global__ void finalize_kernel(const float* __restrict__ accum, float* __restrict__ out) {
  if (blockIdx.x == 0 && threadIdx.x == 0) {
    const float sum = accum[0], cnt = accum[1];
    out[0] = (cnt > 0.0f) ? (sum / fmaxf(cnt, 1.0f)) : 0.0f;
  }
}

extern "C" void kernel_launch(void* const* d_in, const int* in_sizes, int n_in,
                              void* d_out, int out_size, void* d_ws, size_t ws_size,
                              hipStream_t stream) {
  const float* emb    = (const float*)d_in[0];
  const int*   labels = (const int*)d_in[1];
  float*       out    = (float*)d_out;

  char* ws = (char*)d_ws;
  _Float16* xh    = (_Float16*)ws;                                    // 2 MB f16 matrix
  float*    sq    = (float*)(ws + (size_t)N_SAMP * DIMF * 2);         // 32 KB
  float*    accum = (float*)(ws + (size_t)N_SAMP * DIMF * 2 + (size_t)N_SAMP * 4);

  normalize_kernel<<<N_SAMP / 8, 256, 0, stream>>>(emb, xh, sq, accum);
  triplet_tiles_kernel<<<N_SAMP / 16, 256, 0, stream>>>(xh, sq, labels, accum);
  finalize_kernel<<<1, 32, 0, stream>>>(accum, out);
}
